// GCNEncoder_63204738728398
// MI455X (gfx1250) — compile-verified, matrix operand order checked by reading, not run on hardware
//
#include <hip/hip_runtime.h>
#include <hip/hip_bf16.h>

#define NNODES 100000
#define NEDGES 3200000
#define NF 512
#define NH 256
#define NC 40
#define NCP 48   // NC padded to 3 WMMA tiles

typedef __bf16 bf16x16 __attribute__((ext_vector_type(16)));
typedef float f32x8   __attribute__((ext_vector_type(8)));

union Frag { uint4 u[2]; bf16x16 v; };

__device__ __forceinline__ unsigned short f2bf_raw(float f) {
  __hip_bfloat16 b = __float2bfloat16(f);
  union { __hip_bfloat16 h; unsigned short u; } cv;
  cv.h = b;
  return cv.u;
}

// ---------------- degree / dinv ----------------
__global__ void k_deg_init(float* deg, int n) {
  int i = blockIdx.x * blockDim.x + threadIdx.x;
  if (i < n) deg[i] = 1.0f;                       // self-loop
}

__global__ void k_deg_edge(const int* __restrict__ dst, float* deg, int e) {
  int i = blockIdx.x * blockDim.x + threadIdx.x;
  if (i < e) atomicAdd(&deg[dst[i]], 1.0f);
}

__global__ void k_dinv(float* deg, int n) {
  int i = blockIdx.x * blockDim.x + threadIdx.x;
  if (i < n) deg[i] = rsqrtf(deg[i]);             // in place: deg -> dinv
}

// ---------------- conversions (4 elems/thread, 8B stores) ----------------
__global__ void k_cvt_x4(const float4* __restrict__ x, ushort4* __restrict__ xb, long n4) {
  long i = (long)blockIdx.x * blockDim.x + threadIdx.x;
  if (i < n4) {
    float4 v = x[i];
    ushort4 o;
    o.x = f2bf_raw(v.x); o.y = f2bf_raw(v.y);
    o.z = f2bf_raw(v.z); o.w = f2bf_raw(v.w);
    xb[i] = o;
  }
}

// W1 [NF][NH] row-major  ->  w1t [NH][NF]  (column-major W1, contiguous K per col)
__global__ void k_cvt_w1t(const float* __restrict__ W1, __hip_bfloat16* __restrict__ w1t) {
  int i = blockIdx.x * blockDim.x + threadIdx.x;  // over NH*NF
  if (i < NH * NF) {
    int col = i >> 9, k = i & (NF - 1);
    w1t[i] = __float2bfloat16(W1[k * NH + col]);
  }
}

// W2 [NH][NC] -> w2t [NCP][NH], zero-padded cols 40..47
__global__ void k_cvt_w2t(const float* __restrict__ W2, __hip_bfloat16* __restrict__ w2t) {
  int i = blockIdx.x * blockDim.x + threadIdx.x;  // over NCP*NH
  if (i < NCP * NH) {
    int col = i >> 8, k = i & (NH - 1);
    float v = (col < NC) ? W2[k * NC + col] : 0.0f;
    w2t[i] = __float2bfloat16(v);
  }
}

// ---------------- WMMA GEMM:  C[M x ldc] = A[M x K] * BT[ntiles*16 x K]^T ----------
// grid.x = M/16; each wave computes TWO adjacent 16x16 n-tiles from one A fragment.
template <int K>
__global__ void k_wmma_gemm(const __hip_bfloat16* __restrict__ A,
                            const __hip_bfloat16* __restrict__ BT,
                            float* __restrict__ Cmat, int ldc, int ntiles) {
  const int wave = threadIdx.x >> 5;
  const int lane = threadIdx.x & 31;
  const int nt0 = (blockIdx.y * (blockDim.x >> 5) + wave) * 2;
  if (nt0 >= ntiles) return;
  const bool has1 = (nt0 + 1) < ntiles;
  const int m0 = blockIdx.x << 4;
  const int r = lane & 15;
  const int half = lane >> 4;

  // A frag (16x32 bf16): lane<16 holds row r, K {k0+0..7, k0+16..23}; lane>=16 holds {+8..15, +24..31}
  const __hip_bfloat16* Arow  = A + (long)(m0 + r) * K + half * 8;
  const __hip_bfloat16* Brow0 = BT + (long)(nt0 * 16 + r) * K + half * 8;
  const __hip_bfloat16* Brow1 = has1 ? (Brow0 + 16 * K) : Brow0;  // alias if no 2nd tile

  f32x8 acc0 = {}, acc1 = {};
#pragma unroll
  for (int k0 = 0; k0 < K; k0 += 32) {
    Frag fa, fb0, fb1;
    fa.u[0]  = *(const uint4*)(Arow + k0);
    fa.u[1]  = *(const uint4*)(Arow + k0 + 16);
    fb0.u[0] = *(const uint4*)(Brow0 + k0);
    fb0.u[1] = *(const uint4*)(Brow0 + k0 + 16);
    fb1.u[0] = *(const uint4*)(Brow1 + k0);
    fb1.u[1] = *(const uint4*)(Brow1 + k0 + 16);
    acc0 = __builtin_amdgcn_wmma_f32_16x16x32_bf16(false, fa.v, false, fb0.v,
                                                   (short)0, acc0, false, false);
    acc1 = __builtin_amdgcn_wmma_f32_16x16x32_bf16(false, fa.v, false, fb1.v,
                                                   (short)0, acc1, false, false);
  }

  // C/D layout: VGPR i -> row m0 + half*8 + i, col n0 + r
  float* Crow0 = Cmat + (long)(m0 + half * 8) * ldc + nt0 * 16 + r;
#pragma unroll
  for (int i = 0; i < 8; ++i) Crow0[(long)i * ldc] = acc0[i];
  if (has1) {
    float* Crow1 = Crow0 + 16;
#pragma unroll
    for (int i = 0; i < 8; ++i) Crow1[(long)i * ldc] = acc1[i];
  }
}

// ---------------- aggregation layer 1 (width 256) ----------------
__global__ void k_agg1_init(const float* __restrict__ h, const float* __restrict__ dinv,
                            const float* __restrict__ b1, float* __restrict__ agg, long total) {
  long t = (long)blockIdx.x * blockDim.x + threadIdx.x;
  if (t < total) {
    int i = (int)(t >> 8);
    int j = (int)(t & (NH - 1));
    float di = dinv[i];
    agg[t] = b1[j] + h[t] * di * di;              // bias + self-loop term
  }
}

// one wave per edge: 32 lanes x 2 float4 = 256 features; edge metadata loaded once/wave
__global__ void k_agg1_edge(const float* __restrict__ h, const float* __restrict__ dinv,
                            const int* __restrict__ src, const int* __restrict__ dst,
                            float* agg, int E) {
  int wave = threadIdx.x >> 5;
  int lane = threadIdx.x & 31;
  int e = blockIdx.x * (blockDim.x >> 5) + wave;
  if (e >= E) return;
  int s = src[e], d = dst[e];                     // uniform within the wave
  float norm = dinv[s] * dinv[d];
  const float4* hp = (const float4*)(h + (long)s * NH);
  float* o = agg + (long)d * NH;
  float4 v0 = hp[lane];
  float4 v1 = hp[lane + 32];
  int c0 = lane * 4, c1 = (lane + 32) * 4;
  atomicAdd(o + c0 + 0, v0.x * norm);
  atomicAdd(o + c0 + 1, v0.y * norm);
  atomicAdd(o + c0 + 2, v0.z * norm);
  atomicAdd(o + c0 + 3, v0.w * norm);
  atomicAdd(o + c1 + 0, v1.x * norm);
  atomicAdd(o + c1 + 1, v1.y * norm);
  atomicAdd(o + c1 + 2, v1.z * norm);
  atomicAdd(o + c1 + 3, v1.w * norm);
}

__global__ void k_relu_cvt4(const float4* __restrict__ agg, ushort4* __restrict__ hb, long n4) {
  long t = (long)blockIdx.x * blockDim.x + threadIdx.x;
  if (t < n4) {
    float4 v = agg[t];
    ushort4 o;
    o.x = f2bf_raw(fmaxf(v.x, 0.0f)); o.y = f2bf_raw(fmaxf(v.y, 0.0f));
    o.z = f2bf_raw(fmaxf(v.z, 0.0f)); o.w = f2bf_raw(fmaxf(v.w, 0.0f));
    hb[t] = o;
  }
}

// ---------------- aggregation layer 2 (width 40, h2 stride 48) ----------------
__global__ void k_agg2_init(const float* __restrict__ h2, const float* __restrict__ dinv,
                            const float* __restrict__ b2, float* __restrict__ out, long total) {
  long t = (long)blockIdx.x * blockDim.x + threadIdx.x;
  if (t < total) {                                // total = N * NC
    int i = (int)(t / NC);
    int c = (int)(t % NC);
    float di = dinv[i];
    out[t] = b2[c] + h2[(long)i * NCP + c] * di * di;
  }
}

__global__ void k_agg2_edge(const float* __restrict__ h2, const float* __restrict__ dinv,
                            const int* __restrict__ src, const int* __restrict__ dst,
                            float* out, long total) {   // total = E * 10 (float4 per thread)
  long t = (long)blockIdx.x * blockDim.x + threadIdx.x;
  if (t < total) {
    int e = (int)(t / 10);
    int q = (int)(t % 10);
    int s = src[e], d = dst[e];
    float norm = dinv[s] * dinv[d];
    const float4 v = *(const float4*)(h2 + (long)s * NCP + q * 4);
    float* o = out + (long)d * NC + q * 4;
    atomicAdd(o + 0, v.x * norm);
    atomicAdd(o + 1, v.y * norm);
    atomicAdd(o + 2, v.z * norm);
    atomicAdd(o + 3, v.w * norm);
  }
}

// ---------------- host ----------------
static inline int cdiv(long a, long b) { return (int)((a + b - 1) / b); }

extern "C" void kernel_launch(void* const* d_in, const int* in_sizes, int n_in,
                              void* d_out, int out_size, void* d_ws, size_t ws_size,
                              hipStream_t stream) {
  const float* x  = (const float*)d_in[0];
  const int* src  = (const int*)d_in[1];
  const int* dst  = (const int*)d_in[2];
  const float* W1 = (const float*)d_in[3];
  const float* b1 = (const float*)d_in[4];
  const float* W2 = (const float*)d_in[5];
  const float* b2 = (const float*)d_in[6];
  float* out = (float*)d_out;

  char* w = (char*)d_ws;
  // workspace layout (with reuse):
  float* dinv           = (float*)(w);                           // N f32 (deg then dinv)
  __hip_bfloat16* w1t   = (__hip_bfloat16*)(w + (1ull << 20));   // 256*512 bf16
  __hip_bfloat16* w2t   = (__hip_bfloat16*)(w + (1ull << 20) + (1ull << 19)); // 48*256 bf16
  __hip_bfloat16* xb    = (__hip_bfloat16*)(w + (2ull << 20));   // N*512 bf16   (102.4 MB)
  float* agg1           = (float*)(w + (2ull << 20));            // reuses xb region after GEMM1
  float* h              = (float*)(w + (106ull << 20));          // N*256 f32    (102.4 MB)
  __hip_bfloat16* h1b   = (__hip_bfloat16*)(w + (106ull << 20)); // reuses h region after agg1
  float* h2             = (float*)(w + (208ull << 20));          // N*48 f32     (19.2 MB)

  const int N = NNODES, E = NEDGES;

  // degrees -> dinv
  k_deg_init<<<cdiv(N, 256), 256, 0, stream>>>(dinv, N);
  k_deg_edge<<<cdiv(E, 256), 256, 0, stream>>>(dst, dinv, E);
  k_dinv<<<cdiv(N, 256), 256, 0, stream>>>(dinv, N);

  // bf16 conversions
  k_cvt_x4<<<cdiv((long)N * NF / 4, 256), 256, 0, stream>>>(
      (const float4*)x, (ushort4*)xb, (long)N * NF / 4);
  k_cvt_w1t<<<cdiv(NH * NF, 256), 256, 0, stream>>>(W1, w1t);
  k_cvt_w2t<<<cdiv(NCP * NH, 256), 256, 0, stream>>>(W2, w2t);

  // GEMM1: h[N x 256] = xb[N x 512] @ W1, WMMA bf16 -> f32 (8 waves x 2 n-tiles)
  {
    dim3 grid(N / 16, 1);
    k_wmma_gemm<NF><<<grid, 256, 0, stream>>>(xb, w1t, h, NH, NH / 16);
  }

  // layer-1 aggregation: bias + self-loop init, wave-per-edge atomics, relu+cvt
  k_agg1_init<<<cdiv((long)N * NH, 256), 256, 0, stream>>>(h, dinv, b1, agg1, (long)N * NH);
  k_agg1_edge<<<cdiv(E, 8), 256, 0, stream>>>(h, dinv, src, dst, agg1, E);
  k_relu_cvt4<<<cdiv((long)N * NH / 4, 256), 256, 0, stream>>>(
      (const float4*)agg1, (ushort4*)h1b, (long)N * NH / 4);

  // GEMM2: h2[N x 48] = h1b[N x 256] @ W2 (padded), WMMA bf16 -> f32 (2 waves: tiles {0,1},{2})
  {
    dim3 grid(N / 16, 1);
    k_wmma_gemm<NH><<<grid, 64, 0, stream>>>(h1b, w2t, h2, NCP, NCP / 16);
  }

  // layer-2 aggregation into d_out [N x 40]
  k_agg2_init<<<cdiv((long)N * NC, 256), 256, 0, stream>>>(h2, dinv, b2, out, (long)N * NC);
  k_agg2_edge<<<cdiv((long)E * 10, 256), 256, 0, stream>>>(h2, dinv, src, dst, out, (long)E * 10);
}
